// EncoderLayer_12661563588896
// MI455X (gfx1250) — matrix operand order repaired and checked by reading, not compiled, not run on hardware
//
#include <hip/hip_runtime.h>
#include <hip/hip_bf16.h>
#include <stdint.h>

// ---------------------------------------------------------------------------
// Encoder layer for MI455X (gfx1250, wave32, WMMA).
// All matmuls run as bf16 x bf16 -> f32 via v_wmma_f32_16x16x32_bf16.
// Attention is flash-style (no [B,H,S,S] materialization).
// GEMM tiles are double-buffered via CDNA5 async global->LDS copies (ASYNCcnt):
// tile t+1 streams into LDS while tile t feeds the WMMA pipes.
// ---------------------------------------------------------------------------

#define B_   4
#define S_   2048
#define D_   1024
#define H_   16
#define DH_  64
#define DFF_ 4096
#define NH_  (H_ * DH_)   // 1024

typedef __attribute__((ext_vector_type(16))) __bf16 v16bf;
typedef __attribute__((ext_vector_type(8)))  float  v8f;

union AFrag { v16bf v; uint32_t u[8]; };
union CAcc  { v8f   v; float    f[8]; };

__device__ __forceinline__ unsigned short f2bf(float f) {
  union { float f; uint32_t u; } x; x.f = f;
  uint32_t u = x.u;
  u += 0x7FFFu + ((u >> 16) & 1u);   // round-to-nearest-even
  return (unsigned short)(u >> 16);
}

// Async global->LDS copy of 16 bytes per lane (CDNA5 GLOBAL_LOAD_ASYNC_TO_LDS_B128).
// LDS byte address = low 32 bits of the generic shared-aperture pointer.
__device__ __forceinline__ void async_cp16(unsigned short* dst_lds,
                                           const unsigned short* src_g) {
  uint32_t lds = (uint32_t)(uintptr_t)dst_lds;
  asm volatile("global_load_async_to_lds_b128 %0, %1, off"
               :
               : "v"(lds), "v"(src_g)
               : "memory");
}
__device__ __forceinline__ void async_wait0() {
  asm volatile("s_wait_asynccnt 0x0" ::: "memory");
}
// Allow 4 outstanding async ops (the in-flight next tile) -- async loads
// complete in order, so this guarantees the previous tile's 4 copies landed.
__device__ __forceinline__ void async_wait4() {
  asm volatile("s_wait_asynccnt 0x4" ::: "memory");
}

// ---------------------------------------------------------------------------
// Elementwise conversion kernels (one-time prep of bf16 operands)
// ---------------------------------------------------------------------------
__global__ void cvt_f32_bf16(const float* __restrict__ in,
                             unsigned short* __restrict__ out, int n) {
  int i = blockIdx.x * blockDim.x + threadIdx.x;
  if (i < n) out[i] = f2bf(in[i]);
}

// Wq [H][D][DH] (f32) -> out [H*DH][D] (bf16), out[(h*DH+e)*D + d] = Wq[h][d][e]
__global__ void cvt_wqkv(const float* __restrict__ w,
                         unsigned short* __restrict__ out) {
  int i = blockIdx.x * blockDim.x + threadIdx.x;
  if (i >= H_ * D_ * DH_) return;
  int e = i % DH_;
  int d = (i / DH_) % D_;
  int h = i / (DH_ * D_);
  out[(size_t)(h * DH_ + e) * D_ + d] = f2bf(w[i]);
}

// in [R][C] f32 -> out [C][R] bf16 (transpose + convert)
__global__ void cvt_transpose(const float* __restrict__ in,
                              unsigned short* __restrict__ out, int R, int C) {
  int i = blockIdx.x * blockDim.x + threadIdx.x;
  if (i >= R * C) return;
  int c = i % C, r = i / C;
  out[(size_t)c * R + r] = f2bf(in[i]);
}

// ---------------------------------------------------------------------------
// Generic bf16 GEMM:  C[M,N] = A[M,K] * Bt[N,K]^T + bias, optional ReLU.
// Block tile 128x128, K-step 32, 8 waves each owning a 32x64 C tile.
// Double-buffered async global->LDS staging.
// ---------------------------------------------------------------------------
#define GBM 128
#define GBN 128
#define GBK 32
#define GLD 48   // LDS row stride (bf16 elems); 96B rows keep 16B alignment

__launch_bounds__(256)
__global__ void gemm_bf16(const unsigned short* __restrict__ A,
                          const unsigned short* __restrict__ Bt,
                          const float* __restrict__ bias,
                          float* __restrict__ Cf,
                          unsigned short* __restrict__ Cb,
                          int M, int N, int K, int relu) {
  __shared__ unsigned short As[2][GBM * GLD];
  __shared__ unsigned short Bs[2][GBN * GLD];

  const int tid  = threadIdx.x;
  const int lane = tid & 31;
  const int wid  = tid >> 5;
  const int half = lane >> 4;
  const int l16  = lane & 15;

  const int bm = blockIdx.y * GBM;
  const int bn = blockIdx.x * GBN;
  const int wm = (wid >> 1) * 32;   // 4 waves along M
  const int wn = (wid & 1) * 64;    // 2 waves along N

  CAcc acc[2][4];
#pragma unroll
  for (int mi = 0; mi < 2; ++mi)
#pragma unroll
    for (int ni = 0; ni < 4; ++ni)
#pragma unroll
      for (int j = 0; j < 8; ++j) acc[mi][ni].f[j] = 0.f;

  const int lrow = tid >> 1;          // 0..127
  const int lcol = (tid & 1) * 16;    // 0 / 16

  // per-thread staging: 4 async 16B copies per tile
  auto stage = [&](int buf, int kk) {
    const unsigned short* gA = A + (size_t)(bm + lrow) * K + kk + lcol;
    unsigned short* sA = &As[buf][lrow * GLD + lcol];
    async_cp16(sA, gA);
    async_cp16(sA + 8, gA + 8);
    const unsigned short* gB = Bt + (size_t)(bn + lrow) * K + kk + lcol;
    unsigned short* sB = &Bs[buf][lrow * GLD + lcol];
    async_cp16(sB, gB);
    async_cp16(sB + 8, gB + 8);
  };

  stage(0, 0);
  int cur = 0;

  for (int k0 = 0; k0 < K; k0 += GBK) {
    // prefetch next tile into the other buffer, then wait for current tile
    if (k0 + GBK < K) {
      stage(cur ^ 1, k0 + GBK);
      async_wait4();            // current tile's 4 copies have landed
    } else {
      async_wait0();
    }
    __syncthreads();

    const unsigned short* Ac = As[cur];
    const unsigned short* Bc = Bs[cur];

    // ---- gather WMMA fragments (wave32 layouts) ----
    AFrag af[2];
#pragma unroll
    for (int mi = 0; mi < 2; ++mi) {
      const unsigned short* base = Ac + (wm + mi * 16 + l16) * GLD;
#pragma unroll
      for (int j = 0; j < 8; ++j) {
        int kk = (half << 4) + (j << 1);
        af[mi].u[j] = *(const uint32_t*)(base + kk);
      }
    }
    AFrag bf[4];
#pragma unroll
    for (int ni = 0; ni < 4; ++ni) {
      const unsigned short* base = Bc + (wn + ni * 16 + l16) * GLD;
#pragma unroll
      for (int j = 0; j < 8; ++j)
        bf[ni].u[j] = *(const uint32_t*)(base + (half << 4) + (j << 1));
    }

#pragma unroll
    for (int mi = 0; mi < 2; ++mi)
#pragma unroll
      for (int ni = 0; ni < 4; ++ni)
        acc[mi][ni].v = __builtin_amdgcn_wmma_f32_16x16x32_bf16(
            false, af[mi].v, false, bf[ni].v, (short)0, acc[mi][ni].v,
            false, false);
    __syncthreads();   // all waves done reading buf[cur] before it is re-staged
    cur ^= 1;
  }

  // ---- epilogue: bias, ReLU, f32 and/or bf16 stores ----
#pragma unroll
  for (int ni = 0; ni < 4; ++ni) {
    const int n = bn + wn + ni * 16 + l16;
    const float bv = bias ? bias[n] : 0.f;
#pragma unroll
    for (int mi = 0; mi < 2; ++mi) {
#pragma unroll
      for (int j = 0; j < 8; ++j) {
        const int m = bm + wm + mi * 16 + (half << 3) + j;
        float v = acc[mi][ni].f[j] + bv;
        if (relu) v = fmaxf(v, 0.f);
        const size_t off = (size_t)m * N + n;
        if (Cf) Cf[off] = v;
        if (Cb) Cb[off] = f2bf(v);
      }
    }
  }
}

// ---------------------------------------------------------------------------
// Flash attention: per (b, h, 128 q-rows) block; 8 waves x 16 q-rows each.
// 64-key tiles staged in LDS; online softmax; P routed through LDS to convert
// the WMMA C layout into an A-operand layout for the P@V matmul.
// ---------------------------------------------------------------------------
#define QTILE 128
#define KTILE 64
#define LQK   72   // LDS row stride for 64-wide rows (144B, 16B-aligned)

__launch_bounds__(256)
__global__ void flash_attn(const unsigned short* __restrict__ Q,
                           const unsigned short* __restrict__ Kb,
                           const unsigned short* __restrict__ Vb,
                           const unsigned char* __restrict__ mask,
                           unsigned short* __restrict__ ctx) {
  __shared__ unsigned short sm[KTILE * LQK + KTILE * LQK + QTILE * LQK];
  unsigned short* ks = sm;                       // [key][dh]
  unsigned short* vs = sm + KTILE * LQK;         // [dh][key] (transposed)
  unsigned short* ps = sm + 2 * KTILE * LQK;     // per-wave P tiles; doubles as qs

  const int tid  = threadIdx.x;
  const int lane = tid & 31;
  const int w    = tid >> 5;
  const int half = lane >> 4;
  const int l16  = lane & 15;
  const int b  = blockIdx.z;
  const int h  = blockIdx.y;
  const int q0 = blockIdx.x * QTILE;

  // ---- stage the 128x64 Q tile (async, reusing the ps region as qs) ----
  {
    const int row = tid >> 1;
    const int col = (tid & 1) * 32;
    const unsigned short* g = Q + (size_t)(b * S_ + q0 + row) * NH_ + h * DH_ + col;
    unsigned short* s = ps + row * LQK + col;
    async_cp16(s, g);
    async_cp16(s + 8, g + 8);
    async_cp16(s + 16, g + 16);
    async_cp16(s + 24, g + 24);
  }
  async_wait0();
  __syncthreads();

  AFrag aq[2];
  {
    const unsigned short* base = ps + (w * 16 + l16) * LQK;
#pragma unroll
    for (int c = 0; c < 2; ++c)
#pragma unroll
      for (int j = 0; j < 8; ++j) {
        int k = c * 32 + (half << 4) + (j << 1);
        aq[c].u[j] = *(const uint32_t*)(base + k);
      }
  }
  __syncthreads();

  unsigned short* psw = ps + w * 16 * LQK;

  CAcc accC[4];
#pragma unroll
  for (int di = 0; di < 4; ++di)
#pragma unroll
    for (int j = 0; j < 8; ++j) accC[di].f[j] = 0.f;

  float mrow[8], lsum[8];
#pragma unroll
  for (int j = 0; j < 8; ++j) { mrow[j] = -3.0e38f; lsum[j] = 0.f; }

  const float SCALE = 0.022097086912079608f;   // 1/sqrt(S) per reference
  const size_t mbase = ((size_t)b * S_ + (q0 + w * 16 + half * 8)) * S_;

  for (int t = 0; t < S_ / KTILE; ++t) {
    const int kb0 = t * KTILE;

    // ---- stage K [key][dh] (async) and V transposed [dh][key] ----
    {
      const int row = tid >> 2;            // key 0..63
      const int col = (tid & 3) * 16;      // dh chunk
      const unsigned short* gk = Kb + (size_t)(b * S_ + kb0 + row) * NH_ + h * DH_ + col;
      unsigned short* sk = ks + row * LQK + col;
      async_cp16(sk, gk);
      async_cp16(sk + 8, gk + 8);
      const uint4* gv = (const uint4*)(Vb + (size_t)(b * S_ + kb0 + row) * NH_ + h * DH_ + col);
      uint4 v0 = gv[0], v1 = gv[1];
      const unsigned short* p0 = (const unsigned short*)&v0;
      const unsigned short* p1 = (const unsigned short*)&v1;
#pragma unroll
      for (int e = 0; e < 8; ++e) vs[(col + e) * LQK + row] = p0[e];
#pragma unroll
      for (int e = 0; e < 8; ++e) vs[(col + 8 + e) * LQK + row] = p1[e];
    }
    async_wait0();
    __syncthreads();

    // ---- scores: S = Q Kt  (16x64 per wave) ----
    CAcc sf[4];
#pragma unroll
    for (int ni = 0; ni < 4; ++ni)
#pragma unroll
      for (int j = 0; j < 8; ++j) sf[ni].f[j] = 0.f;
#pragma unroll
    for (int c = 0; c < 2; ++c) {
#pragma unroll
      for (int ni = 0; ni < 4; ++ni) {
        AFrag bk;
        const unsigned short* base = ks + (ni * 16 + l16) * LQK + c * 32;
#pragma unroll
        for (int j = 0; j < 8; ++j)
          bk.u[j] = *(const uint32_t*)(base + (half << 4) + (j << 1));
        sf[ni].v = __builtin_amdgcn_wmma_f32_16x16x32_bf16(
            false, aq[c].v, false, bk.v, (short)0, sf[ni].v, false, false);
      }
    }

    // ---- scale + mask ----
    float sv[4][8];
#pragma unroll
    for (int ni = 0; ni < 4; ++ni) {
      const int key = kb0 + ni * 16 + l16;
#pragma unroll
      for (int j = 0; j < 8; ++j) {
        float s = sf[ni].f[j] * SCALE;
        if (mask[mbase + (size_t)j * S_ + key]) s = -1.0e9f;
        sv[ni][j] = s;
      }
    }

    // ---- online softmax (row reductions across 16-lane groups) ----
#pragma unroll
    for (int j = 0; j < 8; ++j) {
      float tmax = fmaxf(fmaxf(sv[0][j], sv[1][j]), fmaxf(sv[2][j], sv[3][j]));
#pragma unroll
      for (int off = 8; off >= 1; off >>= 1)
        tmax = fmaxf(tmax, __shfl_xor(tmax, off, 32));
      const float newm = fmaxf(mrow[j], tmax);
      const float corr = __expf(mrow[j] - newm);
      mrow[j] = newm;
      float psum = 0.f;
#pragma unroll
      for (int ni = 0; ni < 4; ++ni) {
        const float p = __expf(sv[ni][j] - newm);
        sv[ni][j] = p;
        psum += p;
      }
#pragma unroll
      for (int off = 8; off >= 1; off >>= 1)
        psum += __shfl_xor(psum, off, 32);
      lsum[j] = lsum[j] * corr + psum;
#pragma unroll
      for (int di = 0; di < 4; ++di) accC[di].f[j] *= corr;
    }

    // ---- write P (bf16) to this wave's LDS area, C-layout -> row-major ----
#pragma unroll
    for (int ni = 0; ni < 4; ++ni)
#pragma unroll
      for (int j = 0; j < 8; ++j)
        psw[(half * 8 + j) * LQK + ni * 16 + l16] = f2bf(sv[ni][j]);
    asm volatile("s_wait_dscnt 0" ::: "memory");   // cross-lane LDS RAW within wave

    // ---- ctx += P @ V ----
#pragma unroll
    for (int c = 0; c < 2; ++c) {
      AFrag ap;
      const unsigned short* base = psw + l16 * LQK + c * 32;
#pragma unroll
      for (int j = 0; j < 8; ++j)
        ap.u[j] = *(const uint32_t*)(base + (half << 4) + (j << 1));
#pragma unroll
      for (int di = 0; di < 4; ++di) {
        AFrag bv;
        const unsigned short* vb2 = vs + (di * 16 + l16) * LQK + c * 32;
#pragma unroll
        for (int j = 0; j < 8; ++j)
          bv.u[j] = *(const uint32_t*)(vb2 + (half << 4) + (j << 1));
        accC[di].v = __builtin_amdgcn_wmma_f32_16x16x32_bf16(
            false, ap.v, false, bv.v, (short)0, accC[di].v, false, false);
      }
    }
    __syncthreads();
  }

  // ---- normalize + store ctx as bf16 [B,S,H*DH] ----
#pragma unroll
  for (int j = 0; j < 8; ++j) {
    const float inv = 1.0f / lsum[j];
    const size_t row =
        (size_t)(b * S_ + q0 + w * 16 + half * 8 + j) * NH_ + h * DH_;
#pragma unroll
    for (int di = 0; di < 4; ++di)
      ctx[row + di * 16 + l16] = f2bf(accC[di].f[j] * inv);
  }
}

// ---------------------------------------------------------------------------
// Host-side orchestration
// ---------------------------------------------------------------------------
extern "C" void kernel_launch(void* const* d_in, const int* in_sizes, int n_in,
                              void* d_out, int out_size, void* d_ws, size_t ws_size,
                              hipStream_t stream) {
  const float* x    = (const float*)d_in[0];
  const unsigned char* mask = (const unsigned char*)d_in[1];
  const float* Wq = (const float*)d_in[2];
  const float* bq = (const float*)d_in[3];
  const float* Wk = (const float*)d_in[4];
  const float* bk = (const float*)d_in[5];
  const float* Wv = (const float*)d_in[6];
  const float* bv = (const float*)d_in[7];
  const float* Wo = (const float*)d_in[8];
  const float* bo = (const float*)d_in[9];
  const float* W1 = (const float*)d_in[10];
  const float* b1 = (const float*)d_in[11];
  const float* W2 = (const float*)d_in[12];
  const float* b2 = (const float*)d_in[13];
  float* out = (float*)d_out;

  char* ws = (char*)d_ws;
  const size_t MB = 1024u * 1024u;
  unsigned short* xb   = (unsigned short*)(ws + 0);        // 16MB  x bf16
  unsigned short* qwt  = (unsigned short*)(ws + 16 * MB);  // 2MB   Wq^T
  unsigned short* kwt  = (unsigned short*)(ws + 18 * MB);
  unsigned short* vwt  = (unsigned short*)(ws + 20 * MB);
  unsigned short* wot  = (unsigned short*)(ws + 22 * MB);  // 2MB
  unsigned short* w1t  = (unsigned short*)(ws + 24 * MB);  // 8MB
  unsigned short* w2t  = (unsigned short*)(ws + 32 * MB);  // 8MB
  unsigned short* qbuf = (unsigned short*)(ws + 40 * MB);  // 16MB
  unsigned short* kbuf = (unsigned short*)(ws + 56 * MB);  // 16MB
  unsigned short* vbuf = (unsigned short*)(ws + 72 * MB);  // 16MB
  unsigned short* ctxb = (unsigned short*)(ws + 88 * MB);  // 16MB
  unsigned short* hbuf = (unsigned short*)(ws + 40 * MB);  // 64MB, aliases q/k/v/ctx (dead)
  unsigned short* aob  = (unsigned short*)(ws + 104 * MB); // 16MB attn-out bf16

  const int M = B_ * S_;   // 8192
  dim3 blk(256);

  // bf16 conversions
  {
    int n = M * D_;
    cvt_f32_bf16<<<(n + 255) / 256, blk, 0, stream>>>(x, xb, n);
    int nw = H_ * D_ * DH_;
    cvt_wqkv<<<(nw + 255) / 256, blk, 0, stream>>>(Wq, qwt);
    cvt_wqkv<<<(nw + 255) / 256, blk, 0, stream>>>(Wk, kwt);
    cvt_wqkv<<<(nw + 255) / 256, blk, 0, stream>>>(Wv, vwt);
    int no = NH_ * D_;
    cvt_transpose<<<(no + 255) / 256, blk, 0, stream>>>(Wo, wot, NH_, D_);
    int n1 = D_ * DFF_;
    cvt_transpose<<<(n1 + 255) / 256, blk, 0, stream>>>(W1, w1t, D_, DFF_);
    cvt_transpose<<<(n1 + 255) / 256, blk, 0, stream>>>(W2, w2t, DFF_, D_);
  }

  // QKV projections: [8192,1024] x [1024,1024]
  dim3 g_qkv(D_ / GBN, M / GBM);
  gemm_bf16<<<g_qkv, blk, 0, stream>>>(xb, qwt, bq, nullptr, qbuf, M, NH_, D_, 0);
  gemm_bf16<<<g_qkv, blk, 0, stream>>>(xb, kwt, bk, nullptr, kbuf, M, NH_, D_, 0);
  gemm_bf16<<<g_qkv, blk, 0, stream>>>(xb, vwt, bv, nullptr, vbuf, M, NH_, D_, 0);

  // Flash attention
  dim3 g_fa(S_ / QTILE, H_, B_);
  flash_attn<<<g_fa, blk, 0, stream>>>(qbuf, kbuf, vbuf, mask, ctxb);

  // Output projection
  gemm_bf16<<<g_qkv, blk, 0, stream>>>(ctxb, wot, bo, nullptr, aob, M, D_, NH_, 0);

  // FFN
  dim3 g_f1(DFF_ / GBN, M / GBM);
  gemm_bf16<<<g_f1, blk, 0, stream>>>(aob, w1t, b1, nullptr, hbuf, M, DFF_, D_, 1);
  dim3 g_f2(D_ / GBN, M / GBM);
  gemm_bf16<<<g_f2, blk, 0, stream>>>(hbuf, w2t, b2, out, nullptr, M, D_, DFF_, 0);

  (void)in_sizes; (void)n_in; (void)out_size; (void)ws_size;
}
